// GraphNetWithSAGPooling_4741643895063
// MI455X (gfx1250) — matrix operand order repaired and checked by reading, not compiled, not run on hardware
//
#include <hip/hip_runtime.h>
#include <hip/hip_bf16.h>

// ---------------------------------------------------------------------------
// GraphNet with SAGPooling for MI455X (gfx1250, wave32, WMMA).
//  - GEMMs: v_wmma_f32_16x16x32_bf16, B matrices pre-packed to WMMA layout
//  - Edge phases: L2-resident f32 atomics (xw2 = 102MB fits 192MB L2)
//  - Top-k staging: global_load_async_to_lds_b32 + s_wait_asynccnt
// ---------------------------------------------------------------------------

#define N_NODES   100000
#define N_EDGES   3200000
#define N_GRAPHS  100
#define N_PER     1000
#define TOPK      800
#define F_IN      128
#define F_MID     16
#define F_OUT     256

typedef __attribute__((ext_vector_type(16))) __bf16 v16bf;
typedef __attribute__((ext_vector_type(8)))  float  v8f;

// -------------------------------- utility ----------------------------------

__global__ void k_zero(float* __restrict__ p, size_t n) {
  size_t i = (size_t)blockIdx.x * blockDim.x + threadIdx.x;
  size_t stride = (size_t)gridDim.x * blockDim.x;
  for (; i < n; i += stride) p[i] = 0.0f;
}

// Pack W1/W2 into bf16 WMMA B-operand register images:
//   wB1[kt][lane][j] = W1[kt*32 + (lane>>4)*8 + (j&7) + (j>>3)*16][lane&15]  (4 K-tiles)
//   wB2[ct][lane][j] = j<8 ? W2[(lane>>4)*8 + j][ct*16 + (lane&15)] : 0     (16 N-tiles,
//                                                K=16 zero-padded to 32)
__global__ void k_prep_w(const float* __restrict__ W1, const float* __restrict__ W2,
                         __bf16* __restrict__ wB1, __bf16* __restrict__ wB2) {
  int t = threadIdx.x;                           // single block, 256 threads
  for (int idx = t; idx < 4 * 32 * 16; idx += 256) {
    int j = idx & 15, lane = (idx >> 4) & 31, kt = idx >> 9;
    int hl = lane >> 4, n = lane & 15;
    wB1[idx] = (__bf16)W1[(kt * 32 + hl * 8 + (j & 7) + (j >> 3) * 16) * F_MID + n];
  }
  for (int idx = t; idx < 16 * 32 * 16; idx += 256) {
    int j = idx & 15, lane = (idx >> 4) & 31, ct = idx >> 9;
    int hl = lane >> 4, n = lane & 15;
    wB2[idx] = (j < 8) ? (__bf16)W2[(hl * 8 + j) * F_OUT + ct * 16 + n] : (__bf16)0.0f;
  }
}

// -------------------------------- conv1 ------------------------------------

// deg1[d] = number of incoming edges (float)
__global__ void k_deg1(const int* __restrict__ dst, float* __restrict__ deg1) {
  int e = blockIdx.x * blockDim.x + threadIdx.x;
  if (e < N_EDGES) atomicAdd(&deg1[dst[e]], 1.0f);
}

// xw1 = x @ W1   [100000x128]@[128x16] -> [100000x16]
// one wave per 16-row tile; 4 WMMA K-steps of 32 (bf16 in, f32 accumulate)
__global__ void k_gemm1(const float* __restrict__ x, const __bf16* __restrict__ wB1,
                        float* __restrict__ xw1) {
  int wave = (int)((blockIdx.x * (size_t)blockDim.x + threadIdx.x) >> 5);
  if (wave >= N_NODES / 16) return;              // wave-uniform exit
  int lane = threadIdx.x & 31;
  int hl = lane >> 4;                            // lane-half (ISA layout split)
  int mn = lane & 15;                            // A: row, B/C: column
  int rowBase = wave * 16;
  const v16bf* wb = (const v16bf*)wB1;           // one v16bf per (tile, lane)

  v8f c = {};
  for (int kk = 0; kk < F_IN; kk += 32) {
    v16bf a;
    // A 16x32 bf16: lane half 0 -> K = kk + {0..7, 16..23}; half 1 -> +8
    const float* ar = x + (size_t)(rowBase + mn) * F_IN + kk + hl * 8;
#pragma unroll
    for (int j = 0; j < 8; ++j) {
      a[j]     = (__bf16)ar[j];
      a[8 + j] = (__bf16)ar[16 + j];
    }
    v16bf b = wb[(kk >> 5) * 32 + lane];         // packed, contiguous per lane
    c = __builtin_amdgcn_wmma_f32_16x16x32_bf16(false, a, false, b,
                                                (short)0, c, false, false);
  }
  // C/D: VGPR r -> row (r + 8*half), col = lane&15
#pragma unroll
  for (int r = 0; r < 8; ++r)
    xw1[(size_t)(rowBase + r + hl * 8) * F_MID + mn] = c[r];
}

// agg1[dst] += xw1[src] * rsqrt(deg_src)*rsqrt(deg_dst), deg = deg1 + 1 (self loop)
__global__ void k_conv1_scatter(const int* __restrict__ src, const int* __restrict__ dst,
                                const float* __restrict__ deg1,
                                const float* __restrict__ xw1,
                                float* __restrict__ agg1) {
  int e = blockIdx.x * blockDim.x + threadIdx.x;
  if (e >= N_EDGES) return;
  int s = src[e], d = dst[e];
  float coef = rsqrtf(deg1[s] + 1.0f) * rsqrtf(deg1[d] + 1.0f);
  const float* xs = xw1 + (size_t)s * F_MID;
  float* ad = agg1 + (size_t)d * F_MID;
#pragma unroll
  for (int f = 0; f < F_MID; ++f) atomicAdd(ad + f, xs[f] * coef);
}

// h = relu(agg1 + xw1 * 1/(deg1+1) + b1)
__global__ void k_h_finalize(const float* __restrict__ agg1, const float* __restrict__ xw1,
                             const float* __restrict__ deg1, const float* __restrict__ b1,
                             float* __restrict__ h) {
  int idx = blockIdx.x * blockDim.x + threadIdx.x;
  if (idx >= N_NODES * F_MID) return;
  int i = idx >> 4, f = idx & 15;
  float v = agg1[idx] + xw1[idx] / (deg1[i] + 1.0f) + b1[f];
  h[idx] = fmaxf(v, 0.0f);
}

// -------------------------------- SAGPool ----------------------------------

// aggs[dst] += h[src]
__global__ void k_score_scatter(const int* __restrict__ src, const int* __restrict__ dst,
                                const float* __restrict__ h, float* __restrict__ aggs) {
  int e = blockIdx.x * blockDim.x + threadIdx.x;
  if (e >= N_EDGES) return;
  int s = src[e], d = dst[e];
  const float* hs = h + (size_t)s * F_MID;
  float* ad = aggs + (size_t)d * F_MID;
#pragma unroll
  for (int f = 0; f < F_MID; ++f) atomicAdd(ad + f, hs[f]);
}

// score = tanh(aggs @ w_rel + h @ w_root + b_rel)
__global__ void k_score(const float* __restrict__ aggs, const float* __restrict__ h,
                        const float* __restrict__ w_rel, const float* __restrict__ w_root,
                        const float* __restrict__ b_rel, float* __restrict__ score) {
  int i = blockIdx.x * blockDim.x + threadIdx.x;
  if (i >= N_NODES) return;
  float acc = b_rel[0];
#pragma unroll
  for (int f = 0; f < F_MID; ++f)
    acc += aggs[(size_t)i * F_MID + f] * w_rel[f] + h[(size_t)i * F_MID + f] * w_root[f];
  score[i] = tanhf(acc);
}

// per-graph stable top-K mask via rank counting in LDS (ties -> lower index wins).
// LDS staging uses the gfx1250 async-to-LDS path (ASYNCcnt).
__global__ void k_topk(const float* __restrict__ score, float* __restrict__ mask) {
  __shared__ float s[N_PER];
  int g = blockIdx.x, t = threadIdx.x;
  if (t < N_PER) {
    // flat address of a __shared__ object: low 32 bits carry the LDS offset
    unsigned lds_off = (unsigned)(unsigned long long)&s[t];
    unsigned long long gaddr =
        (unsigned long long)(score + (size_t)g * N_PER + t);
    asm volatile("global_load_async_to_lds_b32 %0, %1, off"
                 :: "v"(lds_off), "v"(gaddr) : "memory");
  }
  asm volatile("s_wait_asynccnt 0x0" ::: "memory");
  __syncthreads();
  if (t >= N_PER) return;
  float my = s[t];
  int rank = 0;
  for (int j = 0; j < N_PER; ++j) {              // LDS broadcast reads
    float v = s[j];
    rank += (v > my) || (v == my && j < t);
  }
  mask[(size_t)g * N_PER + t] = (rank < TOPK) ? 1.0f : 0.0f;
}

// h *= score * mask (in place)
__global__ void k_gate(float* __restrict__ h, const float* __restrict__ score,
                       const float* __restrict__ mask) {
  int idx = blockIdx.x * blockDim.x + threadIdx.x;
  if (idx >= N_NODES * F_MID) return;
  int i = idx >> 4;
  h[idx] *= score[i] * mask[i];
}

// -------------------------------- conv2 ------------------------------------

// deg2[d] += mask[s]*mask[d]
__global__ void k_deg2(const int* __restrict__ src, const int* __restrict__ dst,
                       const float* __restrict__ mask, float* __restrict__ deg2) {
  int e = blockIdx.x * blockDim.x + threadIdx.x;
  if (e >= N_EDGES) return;
  int s = src[e], d = dst[e];
  float em = mask[s] * mask[d];
  if (em != 0.0f) atomicAdd(&deg2[d], em);
}

// xw2 = h @ W2   [100000x16]@[16x256] -> [100000x256]
// one wave per 16x16 output tile; K=16 zero-padded to one K=32 WMMA
__global__ void k_gemm2(const float* __restrict__ h, const __bf16* __restrict__ wB2,
                        float* __restrict__ xw2) {
  int wave = (int)((blockIdx.x * (size_t)blockDim.x + threadIdx.x) >> 5);
  if (wave >= (N_NODES / 16) * (F_OUT / 16)) return;   // wave-uniform exit
  int lane = threadIdx.x & 31;
  int hl = lane >> 4;
  int mn = lane & 15;
  int rowBase = (wave >> 4) * 16;
  int colBase = (wave & 15) * 16;

  v16bf a = {};
  // A 16x32: K 0..15 real (elements 0..7 per half), K 16..31 zero pad
  const float* ar = h + (size_t)(rowBase + mn) * F_MID + hl * 8;
#pragma unroll
  for (int j = 0; j < 8; ++j) a[j] = (__bf16)ar[j];
  // packed B tile (zero pad baked in): one contiguous 32B load per lane
  v16bf b = ((const v16bf*)wB2)[(wave & 15) * 32 + lane];
  v8f c = {};
  c = __builtin_amdgcn_wmma_f32_16x16x32_bf16(false, a, false, b,
                                              (short)0, c, false, false);
#pragma unroll
  for (int r = 0; r < 8; ++r)
    xw2[(size_t)(rowBase + r + hl * 8) * F_OUT + colBase + mn] = c[r];
}

// one block (256 lanes) per edge: agg2[dst][f] += xw2[src][f] * coef
__global__ void k_conv2_scatter(const int* __restrict__ src, const int* __restrict__ dst,
                                const float* __restrict__ mask, const float* __restrict__ deg2,
                                const float* __restrict__ xw2, float* __restrict__ agg2) {
  int e = blockIdx.x;
  int s = src[e], d = dst[e];
  float em = mask[s] * mask[d];
  if (em == 0.0f) return;                        // block-uniform
  float ds_ = deg2[s] + mask[s];                 // >= 1 here
  float dd_ = deg2[d] + mask[d];                 // >= 1 here
  float coef = em * rsqrtf(ds_) * rsqrtf(dd_);
  int f = threadIdx.x;
  atomicAdd(agg2 + (size_t)d * F_OUT + f, xw2[(size_t)s * F_OUT + f] * coef);
}

// block per graph, thread per feature: out = mean over kept nodes of
// relu(agg2 + xw2 * mask/deg + b2); kept count is exactly TOPK per graph.
__global__ void k_pool(const float* __restrict__ agg2, const float* __restrict__ xw2,
                       const float* __restrict__ mask, const float* __restrict__ deg2,
                       const float* __restrict__ b2, float* __restrict__ out) {
  int g = blockIdx.x, f = threadIdx.x;           // 256 threads
  float bf = b2[f];
  float acc = 0.0f;
  for (int n = 0; n < N_PER; ++n) {
    int i = g * N_PER + n;
    float mi = mask[i];
    if (mi != 0.0f) {
      size_t idx = (size_t)i * F_OUT + f;
      float degv = deg2[i] + mi;                 // >= 1
      float v = agg2[idx] + xw2[idx] * (mi / degv) + bf;
      acc += fmaxf(v, 0.0f);
    }
  }
  out[(size_t)g * F_OUT + f] = acc * (1.0f / (float)TOPK);
}

// ------------------------------- launcher ----------------------------------

extern "C" void kernel_launch(void* const* d_in, const int* in_sizes, int n_in,
                              void* d_out, int out_size, void* d_ws, size_t ws_size,
                              hipStream_t stream) {
  (void)in_sizes; (void)n_in; (void)out_size; (void)ws_size;

  const float* x      = (const float*)d_in[0];
  const int*   ei     = (const int*)  d_in[1];
  /* batch d_in[2] is i / N_PER by construction */
  const float* W1     = (const float*)d_in[3];
  const float* b1     = (const float*)d_in[4];
  const float* w_rel  = (const float*)d_in[5];
  const float* b_rel  = (const float*)d_in[6];
  const float* w_root = (const float*)d_in[7];
  const float* W2     = (const float*)d_in[8];
  const float* b2     = (const float*)d_in[9];
  const int* src = ei;
  const int* dst = ei + N_EDGES;
  float* out = (float*)d_out;

  // workspace layout (floats); regions needing zero-init come first
  float* ws = (float*)d_ws;
  size_t o = 0;
  float* agg1 = ws + o; o += (size_t)N_NODES * F_MID;
  float* aggs = ws + o; o += (size_t)N_NODES * F_MID;
  float* deg1 = ws + o; o += N_NODES;
  float* deg2 = ws + o; o += N_NODES;
  float* agg2 = ws + o; o += (size_t)N_NODES * F_OUT;
  size_t n_zero = o;                             // everything above starts at 0
  __bf16* wB1 = (__bf16*)(ws + o); o += (4 * 32 * 16) / 2;    // 2048 bf16
  __bf16* wB2 = (__bf16*)(ws + o); o += (16 * 32 * 16) / 2;   // 8192 bf16
  float* xw1   = ws + o; o += (size_t)N_NODES * F_MID;
  float* h     = ws + o; o += (size_t)N_NODES * F_MID;
  float* score = ws + o; o += N_NODES;
  float* maskp = ws + o; o += N_NODES;
  float* xw2   = ws + o; o += (size_t)N_NODES * F_OUT;

  const int B = 256;
  k_zero<<<2048, B, 0, stream>>>(ws, n_zero);
  k_prep_w<<<1, B, 0, stream>>>(W1, W2, wB1, wB2);

  // conv1
  k_deg1<<<N_EDGES / B, B, 0, stream>>>(dst, deg1);
  {
    int waves = N_NODES / 16;                    // 6250
    int blocks = (waves * 32 + B - 1) / B;       // 782
    k_gemm1<<<blocks, B, 0, stream>>>(x, wB1, xw1);
  }
  k_conv1_scatter<<<N_EDGES / B, B, 0, stream>>>(src, dst, deg1, xw1, agg1);
  k_h_finalize<<<(N_NODES * F_MID) / B, B, 0, stream>>>(agg1, xw1, deg1, b1, h);

  // SAGPool
  k_score_scatter<<<N_EDGES / B, B, 0, stream>>>(src, dst, h, aggs);
  k_score<<<(N_NODES + B - 1) / B, B, 0, stream>>>(aggs, h, w_rel, w_root, b_rel, score);
  k_topk<<<N_GRAPHS, 1024, 0, stream>>>(score, maskp);
  k_gate<<<(N_NODES * F_MID) / B, B, 0, stream>>>(h, score, maskp);

  // conv2
  k_deg2<<<N_EDGES / B, B, 0, stream>>>(src, dst, maskp, deg2);
  {
    int waves = (N_NODES / 16) * (F_OUT / 16);   // 100000
    int blocks = (waves * 32 + B - 1) / B;       // 12500
    k_gemm2<<<blocks, B, 0, stream>>>(h, wB2, xw2);
  }
  k_conv2_scatter<<<N_EDGES, B, 0, stream>>>(src, dst, maskp, deg2, xw2, agg2);

  // pooled mean output (no atomics; writes d_out directly)
  k_pool<<<N_GRAPHS, F_OUT, 0, stream>>>(agg2, xw2, maskp, deg2, b2, out);
}
